// InvCrossAttention_15118284882169
// MI455X (gfx1250) — matrix operand order repaired and checked by reading, not compiled, not run on hardware
//
#include <hip/hip_runtime.h>
#include <hip/hip_bf16.h>

// ---------------------------------------------------------------------------
// MI455X (gfx1250) implementation of InvCrossAttention.
// - GEMM-shaped math: v_wmma_f32_16x16x32_bf16 (bf16 A/B, f32 accumulate)
// - A-tile staging: Tensor Data Mover (tensor_load_to_lds + s_wait_tensorcnt)
// - Fragments fetched via ds_load_b128 from conflict-free padded LDS layouts
// - fp32 VALU for softmax / layernorm / norms (precision-critical reductions)
// ---------------------------------------------------------------------------

typedef __bf16 bf16;
typedef bf16  v16bf __attribute__((ext_vector_type(16)));
typedef bf16  v8bf  __attribute__((ext_vector_type(8)));
typedef bf16  v4bf  __attribute__((ext_vector_type(4)));
typedef float v8f   __attribute__((ext_vector_type(8)));
typedef float v4f   __attribute__((ext_vector_type(4)));
typedef unsigned v4u __attribute__((ext_vector_type(4)));
typedef int   v8i   __attribute__((ext_vector_type(8)));
typedef int   v4i   __attribute__((ext_vector_type(4)));

#if __has_builtin(__builtin_amdgcn_tensor_load_to_lds) && \
    __has_builtin(__builtin_amdgcn_s_wait_tensorcnt)
#define HAVE_TDM 1
#endif

#define EPS       1e-6f
#define NEG_SLOPE 0.2f
#define LN_EPS    1e-6f

// problem dims (fixed by the reference)
constexpr int Bb   = 8;      // batch
constexpr int Nn   = 2048;   // sequence
constexpr int Ff   = 3;      // vector axis
constexpr int QKd  = 512;
constexpr int Vd   = 512;
constexpr int Hn   = 8;      // heads
constexpr int Qn   = 64;     // queries
constexpr int DH   = 64;     // head dim

__device__ __forceinline__ bf16 f2bf(float f) {
  // round-to-nearest-even f32 -> bf16
  union { float f; unsigned u; } x; x.f = f;
  unsigned r = (x.u + 0x7fffu + ((x.u >> 16) & 1u)) >> 16;
  union { unsigned short s; bf16 b; } y; y.s = (unsigned short)r;
  return y.b;
}

__device__ __forceinline__ void cvt4(v16bf& d, int base, v4f f) {
  d[base + 0] = f2bf(f[0]); d[base + 1] = f2bf(f[1]);
  d[base + 2] = f2bf(f[2]); d[base + 3] = f2bf(f[3]);
}

__device__ __forceinline__ v16bf join8(v8bf lo, v8bf hi) {
  return __builtin_shufflevector(lo, hi, 0, 1, 2, 3, 4, 5, 6, 7,
                                 8, 9, 10, 11, 12, 13, 14, 15);
}

// ---------------------------------------------------------------------------
// Generic tiled GEMM: C[M,N] = A[M,K] @ B[K,N], row-major, fp32 in/out,
// bf16 WMMA compute. Block = 256 threads = 8 waves; block tile 64x64x32.
// A tile: TDM-staged f32, LDS row stride 36 dwords (pad written by the TDM
//   itself: pad_interval=32 dwords, pad_amount=4 dwords) -> the 16B fragment
//   loads of lanes 0..15 cover all 64 banks exactly once (conflict-free).
// B tile: per-thread transpose+convert into Bt[n][k] bf16 (frag contiguous).
// epi: 0=none, 1=+bias, 2=+bias+leakyrelu.
// ---------------------------------------------------------------------------
__global__ __launch_bounds__(256)
void gemm_wmma(const float* __restrict__ A, const float* __restrict__ B,
               float* __restrict__ C, int M, int N, int K,
               const float* __restrict__ bias, int epi)
{
  __shared__ float Af[64][36];  // 9 KB, f32 A tile (TDM destination, padded)
  __shared__ bf16  Bt[64][40];  // 5 KB, transposed bf16 B tile: Bt[n][k]

  const int tid  = threadIdx.x;
  const int lane = tid & 31;
  const int wave = tid >> 5;
  const int m0   = blockIdx.y * 64;
  const int n0   = blockIdx.x * 64;
  const int tm   = wave >> 1;         // 0..3: 16-row band
  const int tn0  = (wave & 1) * 2;    // 0 or 2: pair of 16-col bands
  const int lm   = lane & 15;
  const int hi   = lane >> 4;

  v8f acc0 = {}; v8f acc1 = {};

  for (int k0 = 0; k0 < K; k0 += 32) {
    // --- A tile: 64x32 f32, moved by the Tensor Data Mover (wave 0 issues) ---
#ifdef HAVE_TDM
    if (wave == 0) {
      unsigned long long ga =
          (unsigned long long)(uintptr_t)(A + (size_t)m0 * K + k0);
      unsigned lds_off = (unsigned)(uintptr_t)(&Af[0][0]);
      // D# group0: count=1 | lds_addr | global_addr[56:0] | type=2
      v4u g0;
      g0[0] = 1u;
      g0[1] = lds_off;
      g0[2] = (unsigned)ga;
      g0[3] = (unsigned)((ga >> 32) & 0x1FFFFFFull) | (2u << 30);
      // D# group1: data_size=4B, pad_enable, pad_interval=32dw, pad_amount=4dw,
      //            tensor_dim0=K, tensor_dim1=M, tile 32x64, stride0=K
      v8i g1;
      g1[0] = (int)((2u << 16) | (1u << 20) | (4u << 22) | (3u << 25));
      g1[1] = (int)(((unsigned)K & 0xFFFFu) << 16);
      g1[2] = (int)(((unsigned)K >> 16) | (((unsigned)M & 0xFFFFu) << 16));
      g1[3] = (int)(((unsigned)M >> 16) | (32u << 16));
      g1[4] = 64;               // tile_dim1 = 64 rows
      g1[5] = (int)(unsigned)K; // tensor_dim0_stride lo32
      g1[6] = 0;
      g1[7] = 0;
      v4i z4 = {0, 0, 0, 0};
#if defined(__clang_major__) && __clang_major__ >= 23
      v8i z8 = {0, 0, 0, 0, 0, 0, 0, 0};
      __builtin_amdgcn_tensor_load_to_lds(g0, g1, z4, z4, z8, 0);
#else
      __builtin_amdgcn_tensor_load_to_lds(g0, g1, z4, z4, 0);
#endif
      __builtin_amdgcn_s_wait_tensorcnt(0);
    }
#else
    // fallback: cooperative b128 copy into the same padded layout
#pragma unroll
    for (int it = 0; it < 2; ++it) {
      int i  = tid + it * 256;
      int r  = i >> 3;           // 0..63
      int c4 = (i & 7) << 2;     // 0..28
      *(v4f*)&Af[r][c4] = *(const v4f*)&A[(size_t)(m0 + r) * K + k0 + c4];
    }
#endif
    // --- B tile, transposed into Bt[n][k]; global reads stay row-coalesced ---
#pragma unroll
    for (int it = 0; it < 2; ++it) {
      int i  = tid + it * 256;
      int c  = i & 63;           // n col
      int r4 = (i >> 6) << 2;    // 0,4,...,28
      const float* bp = &B[(size_t)(k0 + r4) * N + n0 + c];
      v4bf h;
      h[0] = f2bf(bp[0]);
      h[1] = f2bf(bp[(size_t)N]);
      h[2] = f2bf(bp[(size_t)2 * N]);
      h[3] = f2bf(bp[(size_t)3 * N]);
      *(v4bf*)&Bt[c][r4] = h;
    }
    if (k0 + 32 < K) {  // gfx1250 global_prefetch_b8 for the next K-slab
      __builtin_prefetch(&B[(size_t)(k0 + 32 + (tid >> 3)) * N + n0], 0, 1);
    }
    __syncthreads();

    // A fragment (lanes 0-15 K 0-15, lanes 16-31 K 16-31):
    // 4x ds_load_b128 from padded f32 tile, convert to bf16 in VGPRs
    const int ar = tm * 16 + lm;
    const int kb = hi * 16;
    v16bf a;
    cvt4(a, 0,  *(const v4f*)&Af[ar][kb]);
    cvt4(a, 4,  *(const v4f*)&Af[ar][kb + 4]);
    cvt4(a, 8,  *(const v4f*)&Af[ar][kb + 8]);
    cvt4(a, 12, *(const v4f*)&Af[ar][kb + 12]);
    // B fragments (lanes 0-15 K 0-15, lanes 16-31 K 16-31): contiguous in Bt
    const int kb2 = hi * 16;
    v16bf b0 = join8(*(const v8bf*)&Bt[tn0 * 16 + lm][kb2],
                     *(const v8bf*)&Bt[tn0 * 16 + lm][kb2 + 8]);
    v16bf b1 = join8(*(const v8bf*)&Bt[tn0 * 16 + 16 + lm][kb2],
                     *(const v8bf*)&Bt[tn0 * 16 + 16 + lm][kb2 + 8]);

    acc0 = __builtin_amdgcn_wmma_f32_16x16x32_bf16(false, a, false, b0, (short)0, acc0, false, false);
    acc1 = __builtin_amdgcn_wmma_f32_16x16x32_bf16(false, a, false, b1, (short)0, acc1, false, false);
    __syncthreads();
  }

  // C layout: VGPR j -> M = tm*16 + hi*8 + j, col = lane&15 within band
#pragma unroll
  for (int j = 0; j < 8; ++j) {
    int row  = m0 + tm * 16 + hi * 8 + j;
    int col0 = n0 + tn0 * 16 + lm;
    int col1 = col0 + 16;
    float v0 = acc0[j], v1 = acc1[j];
    if (epi) { v0 += bias[col0]; v1 += bias[col1]; }
    if (epi == 2) {
      v0 = v0 > 0.f ? v0 : NEG_SLOPE * v0;
      v1 = v1 > 0.f ? v1 : NEG_SLOPE * v1;
    }
    C[(size_t)row * N + col0] = v0;
    C[(size_t)row * N + col1] = v1;
  }
}

// ---------------------------------------------------------------------------
// EVN helper kernels
// ---------------------------------------------------------------------------
__global__ void evn_norm(const float* __restrict__ pre, float* __restrict__ nrm,
                         int M, int D)
{
  int idx = blockIdx.x * blockDim.x + threadIdx.x;
  if (idx >= M * D) return;
  int m = idx / D, d = idx - m * D;
  const float* p = pre + (size_t)m * Ff * D + d;
  float a = p[0], b = p[D], c = p[2 * D];
  nrm[idx] = sqrtf(a * a + b * b + c * c) + EPS;
}

__global__ void evn_fin(const float* __restrict__ pre, const float* __restrict__ nrm,
                        const float* __restrict__ g, float* __restrict__ out,
                        int M, int D)
{
  int idx = blockIdx.x * blockDim.x + threadIdx.x;
  if (idx >= M * Ff * D) return;
  int d = idx % D;
  int m = idx / (Ff * D);
  float n = nrm[(size_t)m * D + d];
  float x = pre[idx];
  out[idx] = (n <= 2.f * EPS) ? x : x * g[(size_t)m * D + d] / n;
}

// kv_inv[bn,d] = sum_f kv[bn,f,d] * mean_d(kv[bn,f,:])
__global__ __launch_bounds__(256)
void kv_invariant(const float* __restrict__ kv, float* __restrict__ out)
{
  __shared__ float red[256];
  const int bn  = blockIdx.x;
  const int tid = threadIdx.x;
  const float* p = kv + (size_t)bn * Ff * QKd;
  float mean[3];
  for (int f = 0; f < 3; ++f) {
    float s = 0.f;
    for (int d = tid; d < QKd; d += 256) s += p[f * QKd + d];
    red[tid] = s; __syncthreads();
    for (int o = 128; o > 0; o >>= 1) { if (tid < o) red[tid] += red[tid + o]; __syncthreads(); }
    mean[f] = red[0] * (1.f / QKd);
    __syncthreads();
  }
  for (int d = tid; d < QKd; d += 256)
    out[(size_t)bn * QKd + d] =
        p[d] * mean[0] + p[QKd + d] * mean[1] + p[2 * QKd + d] * mean[2];
}

// LayerNorm over contiguous 64-element segments (q/k head vectors), in place
__global__ __launch_bounds__(64)
void layernorm64(float* __restrict__ x, const float* __restrict__ scale,
                 const float* __restrict__ bias)
{
  __shared__ float red[64];
  const int tid = threadIdx.x;
  float* p = x + (size_t)blockIdx.x * 64;
  float v = p[tid];
  red[tid] = v; __syncthreads();
  for (int o = 32; o > 0; o >>= 1) { if (tid < o) red[tid] += red[tid + o]; __syncthreads(); }
  float mean = red[0] * (1.f / 64.f); __syncthreads();
  float dv = v - mean;
  red[tid] = dv * dv; __syncthreads();
  for (int o = 32; o > 0; o >>= 1) { if (tid < o) red[tid] += red[tid + o]; __syncthreads(); }
  float var = red[0] * (1.f / 64.f);
  p[tid] = dv * rsqrtf(var + LN_EPS) * scale[tid] + bias[tid];
}

// ---------------------------------------------------------------------------
// Attention score: (q_ln . k_ln^T)/8 ; wave-per-16x16-tile WMMA, Kdepth=64.
// q layout (Q, H*DH), k layout (B*N, H*DH); attn layout (H*B, Q, N).
// All per-lane gathers are contiguous -> global_load_b128.
// ---------------------------------------------------------------------------
__global__ __launch_bounds__(32)
void attn_score(const float* __restrict__ q, const float* __restrict__ k,
                float* __restrict__ attn)
{
  const int hb = blockIdx.z;      // h*B + b
  const int h  = hb >> 3;
  const int b  = hb & 7;
  const int n0 = blockIdx.x * 16;
  const int q0 = blockIdx.y * 16;
  const int lane = threadIdx.x, lm = lane & 15, hi = lane >> 4;
  const int kb  = hi * 16;

  v8f acc = {};
#pragma unroll
  for (int k0 = 0; k0 < DH; k0 += 32) {
    v16bf av, bv;
    const float* qp = q + (size_t)(q0 + lm) * QKd + h * DH + k0;
    cvt4(av, 0,  *(const v4f*)(qp + kb));
    cvt4(av, 4,  *(const v4f*)(qp + kb + 4));
    cvt4(av, 8,  *(const v4f*)(qp + kb + 8));
    cvt4(av, 12, *(const v4f*)(qp + kb + 12));
    // B[kk][nn] = k[b, n0+nn, h, kk] -> 16 contiguous floats per lane
    const float* kp = k + (size_t)(b * Nn + n0 + lm) * QKd + h * DH + k0 + hi * 16;
    cvt4(bv, 0,  *(const v4f*)(kp));
    cvt4(bv, 4,  *(const v4f*)(kp + 4));
    cvt4(bv, 8,  *(const v4f*)(kp + 8));
    cvt4(bv, 12, *(const v4f*)(kp + 12));
    acc = __builtin_amdgcn_wmma_f32_16x16x32_bf16(false, av, false, bv, (short)0, acc, false, false);
  }
  const float sc = 0.125f;  // 1/sqrt(dh)
#pragma unroll
  for (int j = 0; j < 8; ++j)
    attn[((size_t)hb * Qn + q0 + hi * 8 + j) * Nn + n0 + lm] = acc[j] * sc;
}

// row softmax over N=2048
__global__ __launch_bounds__(256)
void softmax_row(float* __restrict__ attn)
{
  __shared__ float red[256];
  float* p = attn + (size_t)blockIdx.x * Nn;
  const int tid = threadIdx.x;
  float m = -3.4e38f;
  for (int i = tid; i < Nn; i += 256) m = fmaxf(m, p[i]);
  red[tid] = m; __syncthreads();
  for (int o = 128; o > 0; o >>= 1) { if (tid < o) red[tid] = fmaxf(red[tid], red[tid + o]); __syncthreads(); }
  const float mx = red[0]; __syncthreads();
  float s = 0.f;
  for (int i = tid; i < Nn; i += 256) { float e = __expf(p[i] - mx); p[i] = e; s += e; }
  red[tid] = s; __syncthreads();
  for (int o = 128; o > 0; o >>= 1) { if (tid < o) red[tid] += red[tid + o]; __syncthreads(); }
  const float inv = 1.f / red[0];
  for (int i = tid; i < Nn; i += 256) p[i] *= inv;
}

// resi[b,q,f,h*DH+e] = sum_n attn[hb,q,n] * v[b,n,f,h,e]; wave-per-tile, Kdepth=2048
__global__ __launch_bounds__(32)
void attn_apply(const float* __restrict__ attn, const float* __restrict__ v,
                float* __restrict__ out)
{
  const int hb = blockIdx.z, h = hb >> 3, b = hb & 7;
  const int c0 = blockIdx.x * 16;   // column in F*DH = 192
  const int q0 = blockIdx.y * 16;
  const int lane = threadIdx.x, lm = lane & 15, hi = lane >> 4;
  const int c = c0 + lm, f = c >> 6, e = c & 63;
  const int kb = hi * 16;

  v8f acc = {};
  for (int k0 = 0; k0 < Nn; k0 += 32) {
    v16bf av, bv;
    const float* ap = attn + ((size_t)hb * Qn + q0 + lm) * Nn + k0;
    cvt4(av, 0,  *(const v4f*)(ap + kb));
    cvt4(av, 4,  *(const v4f*)(ap + kb + 4));
    cvt4(av, 8,  *(const v4f*)(ap + kb + 8));
    cvt4(av, 12, *(const v4f*)(ap + kb + 12));
#pragma unroll
    for (int i = 0; i < 16; ++i) {   // strided gather, lane-coalesced over e
      int n = k0 + hi * 16 + i;
      bv[i] = f2bf(v[((((size_t)b * Nn + n) * Ff + f) * Hn + h) * DH + e]);
    }
    acc = __builtin_amdgcn_wmma_f32_16x16x32_bf16(false, av, false, bv, (short)0, acc, false, false);
  }
#pragma unroll
  for (int j = 0; j < 8; ++j) {
    int qq = q0 + hi * 8 + j;
    out[(((size_t)b * Qn + qq) * Ff + f) * Vd + h * DH + e] = acc[j];
  }
}

// ---------------------------------------------------------------------------
extern "C" void kernel_launch(void* const* d_in, const int* in_sizes, int n_in,
                              void* d_out, int out_size, void* d_ws, size_t ws_size,
                              hipStream_t stream)
{
  (void)in_sizes; (void)n_in; (void)out_size; (void)ws_size;
  const float* x     = (const float*)d_in[0];
  const float* lq    = (const float*)d_in[1];
  const float* w_kv  = (const float*)d_in[2];
  const float* w_e1a = (const float*)d_in[3];
  const float* b_e1a = (const float*)d_in[4];
  const float* w_e1b = (const float*)d_in[5];
  const float* b_e1b = (const float*)d_in[6];
  const float* wq    = (const float*)d_in[7];
  const float* wk    = (const float*)d_in[8];
  const float* wv    = (const float*)d_in[9];
  const float* lnqs  = (const float*)d_in[10];
  const float* lnqb  = (const float*)d_in[11];
  const float* lnks  = (const float*)d_in[12];
  const float* lnkb  = (const float*)d_in[13];
  const float* w_o   = (const float*)d_in[14];
  const float* w_e2a = (const float*)d_in[15];
  const float* b_e2a = (const float*)d_in[16];
  const float* w_e2b = (const float*)d_in[17];
  const float* b_e2b = (const float*)d_in[18];
  float* out = (float*)d_out;

  float* ws = (float*)d_ws;
  size_t o = 0;
  float* kv   = ws + o; o += (size_t)49152 * 512;   // (B,N,F,QK) pre/post EVN
  float* nrm1 = ws + o; o += (size_t)16384 * 512;
  float* h1   = ws + o; o += (size_t)16384 * 256;
  float* g1   = ws + o; o += (size_t)16384 * 512;
  float* kvin = ws + o; o += (size_t)16384 * 512;
  float* qpre = ws + o; o += (size_t)64 * 512;
  float* kpre = ws + o; o += (size_t)16384 * 512;
  float* vbuf = ws + o; o += (size_t)49152 * 512;   // (B,N,F,H,dv)
  float* attn = ws + o; o += (size_t)4096 * 2048;   // (H,B,Q,N)
  float* r1   = ws + o; o += (size_t)1536 * 512;    // (B,Q,F,V)
  float* r2   = ws + o; o += (size_t)1536 * 512;
  float* nrm2 = ws + o; o += (size_t)512 * 512;
  float* h2   = ws + o; o += (size_t)512 * 256;
  float* g2   = ws + o; o += (size_t)512 * 512;

  // 1. kv_pre = x @ w_kv          (49152 x 256) @ (256 x 512)
  gemm_wmma<<<dim3(8, 768), 256, 0, stream>>>(x, w_kv, kv, 49152, 512, 256, nullptr, 0);
  // 2. EVN nonlinearity 1
  evn_norm<<<(16384 * 512) / 256, 256, 0, stream>>>(kv, nrm1, 16384, 512);
  gemm_wmma<<<dim3(4, 256), 256, 0, stream>>>(nrm1, w_e1a, h1, 16384, 256, 512, b_e1a, 2);
  gemm_wmma<<<dim3(8, 256), 256, 0, stream>>>(h1, w_e1b, g1, 16384, 512, 256, b_e1b, 1);
  evn_fin<<<(16384 * 3 * 512) / 256, 256, 0, stream>>>(kv, nrm1, g1, kv, 16384, 512);
  // 3. invariant contraction
  kv_invariant<<<16384, 256, 0, stream>>>(kv, kvin);
  // 4. q/k/v projections (wq/wk/wv are (512, H*64) row-major matrices)
  gemm_wmma<<<dim3(8, 1), 256, 0, stream>>>(lq, wq, qpre, 64, 512, 512, nullptr, 0);
  gemm_wmma<<<dim3(8, 256), 256, 0, stream>>>(kvin, wk, kpre, 16384, 512, 512, nullptr, 0);
  gemm_wmma<<<dim3(8, 768), 256, 0, stream>>>(kv, wv, vbuf, 49152, 512, 512, nullptr, 0);
  // 5. normalize_qk (LN over each contiguous head vector of 64)
  layernorm64<<<64 * 8, 64, 0, stream>>>(qpre, lnqs, lnqb);
  layernorm64<<<16384 * 8, 64, 0, stream>>>(kpre, lnks, lnkb);
  // 6. attention
  attn_score<<<dim3(128, 4, 64), 32, 0, stream>>>(qpre, kpre, attn);
  softmax_row<<<4096, 256, 0, stream>>>(attn);
  attn_apply<<<dim3(12, 4, 64), 32, 0, stream>>>(attn, vbuf, r1);
  // 7. output projection + EVN nonlinearity 2
  gemm_wmma<<<dim3(8, 24), 256, 0, stream>>>(r1, w_o, r2, 1536, 512, 512, nullptr, 0);
  evn_norm<<<(512 * 512) / 256, 256, 0, stream>>>(r2, nrm2, 512, 512);
  gemm_wmma<<<dim3(4, 8), 256, 0, stream>>>(nrm2, w_e2a, h2, 512, 256, 512, b_e2a, 2);
  gemm_wmma<<<dim3(8, 8), 256, 0, stream>>>(h2, w_e2b, g2, 512, 512, 256, b_e2b, 1);
  evn_fin<<<(512 * 3 * 512) / 256, 256, 0, stream>>>(r2, nrm2, g2, out, 512, 512);
}
